// MVGRL_33732673143022
// MI455X (gfx1250) — matrix operand order repaired and checked by reading, not compiled, not run on hardware
//
#include <hip/hip_runtime.h>

// ---------------------------------------------------------------------------
// MVGRL forward for MI455X (gfx1250, wave32, WMMA).
//
// Cost model: dense GEMMs total ~21.5 GFLOP (negligible on the WMMA pipe);
// the four edge-scatter passes move ~13 GB (largely L2-resident: h is 51 MB
// vs 192 MB L2) and dominate -> scatter uses coalesced float4 row reads +
// f32 global atomics. GEMMs use v_wmma_f32_16x16x32_f16 (f16 in, f32 acc).
// Conv biases b1/b2 cancel exactly under the following BatchNorm and are
// skipped. N=100000 is a multiple of 16, so all M-tiles are full (no row
// predication anywhere near the WMMA loop -> EXEC stays all-ones).
// ---------------------------------------------------------------------------

typedef __attribute__((ext_vector_type(16))) _Float16 v16h;
typedef __attribute__((ext_vector_type(8)))  _Float16 v8h;
typedef __attribute__((ext_vector_type(8)))  float    v8f;

#define HIDD 128
#define BN_EPS 1e-5f

union V16 { v16h v; v8h h[2]; };

// ---------------- utility kernels ----------------

__global__ void zero_f32_kernel(float* __restrict__ p, long long n) {
  long long i = (long long)blockIdx.x * blockDim.x + threadIdx.x;
  if (i < n) p[i] = 0.0f;
}

// f32 -> f16 with optional column padding (cin <= cout; pad cols are zero)
__global__ void cvt_pad_f16_kernel(const float* __restrict__ in, _Float16* __restrict__ out,
                                   int rows, int cin, int cout) {
  long long i = (long long)blockIdx.x * blockDim.x + threadIdx.x;
  long long tot = (long long)rows * cout;
  if (i >= tot) return;
  int r = (int)(i / cout);
  int c = (int)(i % cout);
  out[i] = (c < cin) ? (_Float16)in[(long long)r * cin + c] : (_Float16)0.0f;
}

// ---------------- WMMA GEMM: C[M x ldC] = A[M x K] @ B[K x Nc] (+bias) ------
// A,B f16 row-major; C f32. Wave w of a block computes column-tile w of the
// 16-row M-tile blockIdx.x. B is staged in LDS pre-swizzled into the ISA
// fragment layout Bs[kstep][ntile][lane][16] so each lane fetches its whole
// fragment with two ds_load_b128. A fragment = two contiguous global_load_b128
// per lane per K-step (ISA 16-bit A layout: halfs k0+a_kb..+7, k0+16+a_kb..+7).
// K templated (multiple of 32) -> fully unrolled chain of v_wmma.

template <int K>
__global__ void gemm_wmma_f16_kernel(const _Float16* __restrict__ A,
                                     const _Float16* __restrict__ B,
                                     const float* __restrict__ bias,
                                     float* __restrict__ C,
                                     int Nc, int Nvalid, int ldC) {
  constexpr int KSTEPS = K / 32;
  __shared__ _Float16 Bs[KSTEPS * 8 * 32 * 16];   // up to 64 KB (<< 320 KB/WGP)

  const int tid    = threadIdx.x;
  const int ntiles = Nc >> 4;

  // ---- stage B into fragment layout: element B[k][col] ->
  //      Bs[((kstep*ntiles + (col>>4))*32 + ((col&15) | ((krem&8)<<1)))*16 + ((krem&7) | ((krem&16)>>1))]
  const int tot = K * Nc;
  for (int i = tid; i < tot; i += blockDim.x) {
    int k    = i / Nc;
    int col  = i - k * Nc;
    int ks   = k >> 5;
    int krem = k & 31;
    int lane = (col & 15) | ((krem & 8) << 1);
    int j    = (krem & 7) | ((krem & 16) >> 1);
    Bs[(((ks * ntiles + (col >> 4)) << 5) + lane) * 16 + j] = B[i];
  }
  __syncthreads();

  const int wave = tid >> 5;
  const int lane = tid & 31;
  if (wave >= ntiles) return;                     // wave-uniform

  const int row  = blockIdx.x * 16 + (lane & 15); // full tiles: no guard
  const int col  = wave * 16 + (lane & 15);
  const int a_kb = (lane < 16) ? 0 : 8;
  const _Float16* __restrict__ arow = A + (size_t)row * K;

  v8f acc = {};
#pragma unroll
  for (int ks = 0; ks < KSTEPS; ++ks) {
    V16 a, b;
    a.h[0] = *(const v8h*)(arow + ks * 32 + a_kb);        // global_load_b128
    a.h[1] = *(const v8h*)(arow + ks * 32 + 16 + a_kb);   // global_load_b128
    const _Float16* bp = Bs + (((ks * ntiles + wave) << 5) + lane) * 16;
    b.h[0] = ((const v8h*)bp)[0];                         // ds_load_b128
    b.h[1] = ((const v8h*)bp)[1];                         // ds_load_b128
    acc = __builtin_amdgcn_wmma_f32_16x16x32_f16(
        false, a.v, false, b.v, (short)0, acc, false, false);
  }

  if (col >= Nvalid) return;                      // after all WMMA
  const float bv = bias ? bias[col] : 0.0f;
  const int rbase = blockIdx.x * 16 + ((lane < 16) ? 0 : 8);
#pragma unroll
  for (int r = 0; r < 8; ++r)                     // C/D layout: VGPR r -> row rbase+r
    C[(size_t)(rbase + r) * ldC + col] = acc[r] + bv;
}

// ---------------- edge scatter: agg[dst] += w_e * h[src] -------------------
// 32 lanes per edge, float4 per lane (HID=128). Row reads fully coalesced;
// accumulation via f32 global atomics (working set is L2-resident).

__global__ void scatter_add_kernel(const float* __restrict__ h,
                                   const int* __restrict__ src,
                                   const int* __restrict__ dst,
                                   const float* __restrict__ w,
                                   float* __restrict__ agg, int E) {
  long long gid = (long long)blockIdx.x * blockDim.x + threadIdx.x;
  long long e = gid >> 5;
  int lane = (int)(gid & 31);
  if (e >= E) return;
  int s = src[e], d = dst[e];
  float we = w[e];
  const float4 v = ((const float4*)(h + (size_t)s * HIDD))[lane];
  float* ap = agg + (size_t)d * HIDD + lane * 4;
  atomicAdd(ap + 0, v.x * we);
  atomicAdd(ap + 1, v.y * we);
  atomicAdd(ap + 2, v.z * we);
  atomicAdd(ap + 3, v.w * we);
}

// ---------------- BN column stats: sums[c]=sum, sums[128+c]=sumsq ----------

__global__ void bn_stats_kernel(const float* __restrict__ a, float* __restrict__ sums,
                                int Nrows) {
  const int c  = threadIdx.x & 127;
  const int rr = threadIdx.x >> 7;                // 2 rows per block per iter
  float s = 0.0f, s2 = 0.0f;
  for (long long r = (long long)blockIdx.x * 2 + rr; r < Nrows;
       r += (long long)gridDim.x * 2) {
    float v = a[r * HIDD + c];                    // coalesced across lanes
    s += v; s2 += v * v;
  }
  __shared__ float ls[256], ls2[256];
  ls[threadIdx.x] = s; ls2[threadIdx.x] = s2;
  __syncthreads();
  if (threadIdx.x < 128) {
    atomicAdd(&sums[threadIdx.x],       ls[threadIdx.x]  + ls[threadIdx.x + 128]);
    atomicAdd(&sums[128 + threadIdx.x], ls2[threadIdx.x] + ls2[threadIdx.x + 128]);
  }
}

// ---------------- BN apply (+optional ReLU), write f16 and/or f32(+=) ------

__global__ void bn_apply_kernel(const float* __restrict__ a, const float* __restrict__ sums,
                                const float* __restrict__ gamma, const float* __restrict__ beta,
                                _Float16* __restrict__ out16, float* __restrict__ acc32,
                                int Nrows, int relu, int accumulate) {
  long long i = (long long)blockIdx.x * blockDim.x + threadIdx.x;
  if (i >= (long long)Nrows * HIDD) return;
  int c = (int)(i & (HIDD - 1));
  float inv_n = 1.0f / (float)Nrows;
  float mu  = sums[c] * inv_n;
  float var = sums[128 + c] * inv_n - mu * mu;
  float v = (a[i] - mu) * rsqrtf(var + BN_EPS) * gamma[c] + beta[c];
  if (relu) v = fmaxf(v, 0.0f);
  if (out16) out16[i] = (_Float16)v;
  if (acc32) {
    if (accumulate) acc32[i] += v; else acc32[i] = v;
  }
}

// ---------------------------------------------------------------------------

extern "C" void kernel_launch(void* const* d_in, const int* in_sizes, int n_in,
                              void* d_out, int out_size, void* d_ws, size_t ws_size,
                              hipStream_t stream) {
  const int N = in_sizes[0] / 256;                // 100000 (multiple of 16)
  const int E = in_sizes[1];                      // 3200000
  const int IN_DIM = 256, OUT_DIM = 70, OUT_PAD = 80;

  const float* x    = (const float*)d_in[0];
  const float* ew   = (const float*)d_in[1];
  const float* dwt  = (const float*)d_in[2];
  const int*   eidx = (const int*)d_in[3];        // [2,E]: src then dst
  const int*   didx = (const int*)d_in[4];
  const float* W1a  = (const float*)d_in[5];
  const float* g1a  = (const float*)d_in[7];
  const float* be1a = (const float*)d_in[8];
  const float* W2a  = (const float*)d_in[9];
  const float* g2a  = (const float*)d_in[11];
  const float* be2a = (const float*)d_in[12];
  const float* W1b  = (const float*)d_in[13];
  const float* g1b  = (const float*)d_in[15];
  const float* be1b = (const float*)d_in[16];
  const float* W2b  = (const float*)d_in[17];
  const float* g2b  = (const float*)d_in[19];
  const float* be2b = (const float*)d_in[20];
  const float* Wc   = (const float*)d_in[21];
  const float* bc   = (const float*)d_in[22];
  float* out = (float*)d_out;

  // ---- carve workspace (~256 MB, 256 B aligned) ----
  char* p = (char*)d_ws;
  auto carve = [&](size_t bytes) -> char* {
    char* r = p; p += (bytes + 255) & ~(size_t)255; return r;
  };
  _Float16* x16      = (_Float16*)carve((size_t)N * IN_DIM * 2);
  _Float16* w1a16    = (_Float16*)carve((size_t)IN_DIM * HIDD * 2);
  _Float16* w1b16    = (_Float16*)carve((size_t)IN_DIM * HIDD * 2);
  _Float16* w2a16    = (_Float16*)carve((size_t)HIDD * HIDD * 2);
  _Float16* w2b16    = (_Float16*)carve((size_t)HIDD * HIDD * 2);
  _Float16* wc16     = (_Float16*)carve((size_t)HIDD * OUT_PAD * 2);
  float*    h32      = (float*)carve((size_t)N * HIDD * 4);
  float*    agg      = (float*)carve((size_t)N * HIDD * 4);
  _Float16* h16      = (_Float16*)carve((size_t)N * HIDD * 2);
  float*    encsum   = (float*)carve((size_t)N * HIDD * 4);
  _Float16* encsum16 = (_Float16*)carve((size_t)N * HIDD * 2);
  float*    stats    = (float*)carve(256 * 4);

  const int T = 256;
  auto blocks = [&](long long n) { return (unsigned)((n + T - 1) / T); };
  const int MT = N / 16;                          // 6250 full M-tiles

  // ---- one-time conversions to f16 ----
  cvt_pad_f16_kernel<<<blocks((long long)N * IN_DIM), T, 0, stream>>>(x, x16, N, IN_DIM, IN_DIM);
  cvt_pad_f16_kernel<<<blocks(IN_DIM * HIDD), T, 0, stream>>>(W1a, w1a16, IN_DIM, HIDD, HIDD);
  cvt_pad_f16_kernel<<<blocks(IN_DIM * HIDD), T, 0, stream>>>(W1b, w1b16, IN_DIM, HIDD, HIDD);
  cvt_pad_f16_kernel<<<blocks(HIDD * HIDD), T, 0, stream>>>(W2a, w2a16, HIDD, HIDD, HIDD);
  cvt_pad_f16_kernel<<<blocks(HIDD * HIDD), T, 0, stream>>>(W2b, w2b16, HIDD, HIDD, HIDD);
  cvt_pad_f16_kernel<<<blocks(HIDD * OUT_PAD), T, 0, stream>>>(Wc, wc16, HIDD, OUT_DIM, OUT_PAD);

  const _Float16* W1s[2]  = {w1a16, w1b16};
  const _Float16* W2s[2]  = {w2a16, w2b16};
  const float*    g1s[2]  = {g1a, g1b};
  const float*    be1s[2] = {be1a, be1b};
  const float*    g2s[2]  = {g2a, g2b};
  const float*    be2s[2] = {be2a, be2b};
  const int*      srcs[2] = {eidx, didx};
  const int*      dsts[2] = {eidx + E, didx + E};
  const float*    wgts[2] = {ew, dwt};

  const long long nh = (long long)N * HIDD;
  const unsigned scatterBlocks = blocks((long long)E * 32);

  for (int enc = 0; enc < 2; ++enc) {
    // ---- layer 1: GEMM (x @ W1) -> scatter -> BN -> ReLU (b1 cancels in BN)
    gemm_wmma_f16_kernel<256><<<MT, 8 * 32, 0, stream>>>(
        x16, W1s[enc], nullptr, h32, HIDD, HIDD, HIDD);
    zero_f32_kernel<<<blocks(nh), T, 0, stream>>>(agg, nh);
    zero_f32_kernel<<<1, 256, 0, stream>>>(stats, 256);
    scatter_add_kernel<<<scatterBlocks, T, 0, stream>>>(
        h32, srcs[enc], dsts[enc], wgts[enc], agg, E);
    bn_stats_kernel<<<256, 256, 0, stream>>>(agg, stats, N);
    bn_apply_kernel<<<blocks(nh), T, 0, stream>>>(
        agg, stats, g1s[enc], be1s[enc], h16, nullptr, N, /*relu=*/1, 0);

    // ---- layer 2: GEMM (h @ W2) -> scatter -> BN (no ReLU), sum encoders
    gemm_wmma_f16_kernel<128><<<MT, 8 * 32, 0, stream>>>(
        h16, W2s[enc], nullptr, h32, HIDD, HIDD, HIDD);
    zero_f32_kernel<<<blocks(nh), T, 0, stream>>>(agg, nh);
    zero_f32_kernel<<<1, 256, 0, stream>>>(stats, 256);
    scatter_add_kernel<<<scatterBlocks, T, 0, stream>>>(
        h32, srcs[enc], dsts[enc], wgts[enc], agg, E);
    bn_stats_kernel<<<256, 256, 0, stream>>>(agg, stats, N);
    bn_apply_kernel<<<blocks(nh), T, 0, stream>>>(
        agg, stats, g2s[enc], be2s[enc], nullptr, encsum, N, /*relu=*/0,
        /*accumulate=*/enc);
  }

  // ---- classifier: (enc1+enc2) @ Wc + bc -> d_out [N x 70] ----
  cvt_pad_f16_kernel<<<blocks(nh), T, 0, stream>>>(encsum, encsum16, N, HIDD, HIDD);
  gemm_wmma_f16_kernel<128><<<MT, 5 * 32, 0, stream>>>(
      encsum16, wc16, bc, out, OUT_PAD, OUT_DIM, OUT_DIM);
}
